// GNT_88888643158590
// MI455X (gfx1250) — matrix operand order repaired and checked by reading, not compiled, not run on hardware
//
#include <hip/hip_runtime.h>
#include <hip/hip_bf16.h>

// ---------------------------------------------------------------------------
// Problem dimensions (from the reference)
// ---------------------------------------------------------------------------
#define R_    256
#define S_    64
#define V_    8
#define W_    256
#define D_    4
#define H_    4
#define FEAT_ 35
#define FF_   1024
#define RS_   (R_ * S_)          // 16384
#define RSV_  (R_ * S_ * V_)     // 131072
#define QCW_  (W_ + 126)         // 382 = W + posenc(63) + viewenc(63)
#define QK_   (W_ / H_)          // 64  : q/k projection width
#define HD_   (QK_ / H_)         // 16  : per-head q/k dim
#define VD_   (W_ / H_)          // 64  : per-head v dim

typedef __attribute__((ext_vector_type(16))) _Float16 v16h;
typedef __attribute__((ext_vector_type(8)))  float    v8f;
typedef __attribute__((ext_vector_type(4)))  float    f32x4;
typedef __attribute__((ext_vector_type(4)))  _Float16 h4;
typedef __attribute__((ext_vector_type(8)))  _Float16 h8;

union AFrag { v16h v; h8 h[2]; };

// load 8 consecutive elements as f16x8 (fast path only; 16B aligned)
__device__ __forceinline__ h8 load8_cvt(const float* p)
{
    const f32x4 t0 = *(const f32x4*)p;
    const f32x4 t1 = *(const f32x4*)(p + 4);
    h8 r;
    r[0] = (_Float16)t0[0]; r[1] = (_Float16)t0[1];
    r[2] = (_Float16)t0[2]; r[3] = (_Float16)t0[3];
    r[4] = (_Float16)t1[0]; r[5] = (_Float16)t1[1];
    r[6] = (_Float16)t1[2]; r[7] = (_Float16)t1[3];
    return r;
}
__device__ __forceinline__ h8 load8_cvt(const _Float16* p)
{
    return *(const h8*)(p);
}

// ---------------------------------------------------------------------------
// Generic WMMA GEMM:  C[M,N] = act( A[M,K] @ B[K,N] + bias + resid )
// A: f32 or f16 (TA), B/bias/resid: f32, C: f32 or f16 (TC).
// Block = 256 threads = 8 waves; block tile 64(M) x 64(N); K step 32.
// Waves arranged 2(Mgrp) x 4(N); each wave computes TWO 16x16 WMMA tiles
// stacked in M that SHARE one B fragment (2 WMMA per barrier pair, 6
// ds_load_b128 per 2 WMMAs).
// AL=true  : M%64==0, N%64==0, K%32==0 -> zero guards, vector loads.
// AL=false : index-clamped unconditional loads + cndmask select (no branches).
// LDS rows padded to 40 halves (80B: 16B chunks stay aligned, banks spread).
// B tile stored TRANSPOSED (BsT[n][k]) so fragment gathers are ds_load_b128.
// ---------------------------------------------------------------------------
template <typename TA, typename TC, bool AL>
__global__ void __launch_bounds__(256)
gemm_wmma_kernel(const TA* __restrict__ A, const float* __restrict__ B,
                 const float* __restrict__ bias, const float* __restrict__ resid,
                 TC* __restrict__ C, int M, int N, int K, int relu)
{
    __shared__ _Float16 As[64][40];    // [m][k], row stride 80B (16B aligned)
    __shared__ _Float16 BsT[64][40];   // [n][k], row stride 80B (16B aligned)

    const int tid    = threadIdx.x;
    const int lane   = tid & 31;
    const int wid    = tid >> 5;
    const int wmg    = wid >> 2;        // 0..1  (M group: rows wmg*32 .. +32)
    const int wn     = wid & 3;         // 0..3  (N sub-tile)
    const int laneHi = lane >> 4;       // 0/1
    const int lane16 = lane & 15;
    const int mBase  = blockIdx.y * 64;
    const int nBase  = blockIdx.x * 64;

    // staging coordinates
    const int ar  = tid >> 2;           // A row   0..63
    const int ac0 = (tid & 3) * 8;      // A col   0,8,16,24
    const int bn  = tid & 63;           // B col   0..63
    const int bk0 = (tid >> 6) * 8;     // B k     0,8,16,24

    v8f acc0 = {};
    v8f acc1 = {};

    for (int k0 = 0; k0 < K; k0 += 32) {
        // ---- stage A tile (64 x 32) -> As[m][k] ----
        {
            const int gm = mBase + ar;
            h8 av;
            if (AL) {
                const TA* ap = A + (size_t)gm * K + k0 + ac0;
                av = load8_cvt(ap);
                if (k0 + 32 < K) __builtin_prefetch(ap + 32, 0, 1);   // global_prefetch_b8
            } else {
                const int gmc = gm < M ? gm : M - 1;
#pragma unroll
                for (int j = 0; j < 8; ++j) {
                    const int gk  = k0 + ac0 + j;
                    const int gkc = gk < K ? gk : K - 1;
                    const float t = (float)A[(size_t)gmc * K + gkc];
                    av[j] = (_Float16)((gm < M && gk < K) ? t : 0.0f);
                }
            }
            *(h8*)&As[ar][ac0] = av;                                  // ds_store_b128
        }
        // ---- stage B tile (32 x 64) transposed -> BsT[n][k] ----
        {
            h8 bv;
            if (AL) {
                const float* bp = B + (size_t)(k0 + bk0) * N + nBase + bn;
#pragma unroll
                for (int j = 0; j < 8; ++j) bv[j] = (_Float16)bp[(size_t)j * N];
                if (k0 + 32 < K) __builtin_prefetch(bp + (size_t)32 * N, 0, 1);
            } else {
                const int gn  = nBase + bn;
                const int gnc = gn < N ? gn : N - 1;
#pragma unroll
                for (int j = 0; j < 8; ++j) {
                    const int gk  = k0 + bk0 + j;
                    const int gkc = gk < K ? gk : K - 1;
                    const float t = B[(size_t)gkc * N + gnc];
                    bv[j] = (_Float16)((gk < K && gn < N) ? t : 0.0f);
                }
            }
            *(h8*)&BsT[bn][bk0] = bv;                                 // ds_store_b128
        }
        __syncthreads();

        // ---- WMMA fragments per ISA 7.12.2 wave32 layouts ----
        // A 16x32 f16: lane m=lane%16; halves 0-7 = K[8*laneHi .. +8),
        //              halves 8-15 = K[16+8*laneHi .. +8)  -> two ds_load_b128
        // B 32x16 f16: lane n=lane%16; same K order as A (8*laneHi, 16+8*laneHi)
        AFrag a0, a1, bf;
        const int r0 = wmg * 32 + lane16;
        a0.h[0] = *(const h8*)&As [r0     ][ 8 * laneHi];
        a0.h[1] = *(const h8*)&As [r0     ][16 + 8 * laneHi];
        a1.h[0] = *(const h8*)&As [r0 + 16][ 8 * laneHi];
        a1.h[1] = *(const h8*)&As [r0 + 16][16 + 8 * laneHi];
        bf.h[0] = *(const h8*)&BsT[wn * 16 + lane16][ 8 * laneHi];
        bf.h[1] = *(const h8*)&BsT[wn * 16 + lane16][16 + 8 * laneHi];
        acc0 = __builtin_amdgcn_wmma_f32_16x16x32_f16(
                   false, a0.v, false, bf.v, (short)0, acc0, false, false);
        acc1 = __builtin_amdgcn_wmma_f32_16x16x32_f16(
                   false, a1.v, false, bf.v, (short)0, acc1, false, false);
        __syncthreads();
    }

    // ---- epilogue: C/D layout = lane n=lane%16, vgpr r -> m = 8*laneHi + r ----
    const int gn = nBase + wn * 16 + lane16;
#pragma unroll
    for (int t = 0; t < 2; ++t) {
        const v8f& acc = t ? acc1 : acc0;
#pragma unroll
        for (int r = 0; r < 8; ++r) {
            const int gm = mBase + wmg * 32 + t * 16 + laneHi * 8 + r;
            if (AL || (gm < M && gn < N)) {
                float v = acc[r];
                if (bias)  v += bias[gn];
                if (resid) v += resid[(size_t)gm * N + gn];
                if (relu)  v = fmaxf(v, 0.0f);
                C[(size_t)gm * N + gn] = (TC)v;
            }
        }
    }
}

// ---------------------------------------------------------------------------
// q0 = max over views of feat  ([RS,V,W] f16 -> [RS,W] f32)
// ---------------------------------------------------------------------------
__global__ void rowmax_views_kernel(const _Float16* __restrict__ feat,
                                    float* __restrict__ q)
{
    const int idx = blockIdx.x * blockDim.x + threadIdx.x;
    if (idx >= RS_ * W_) return;
    const int rs = idx / W_, w = idx % W_;
    float m = -3.4e38f;
#pragma unroll
    for (int v = 0; v < V_; ++v)
        m = fmaxf(m, (float)feat[((size_t)rs * V_ + v) * W_ + w]);
    q[idx] = m;
}

// ---------------------------------------------------------------------------
// Row LayerNorm over W=256 (one block per row)
// ---------------------------------------------------------------------------
__global__ void __launch_bounds__(W_)
layernorm_kernel(const float* __restrict__ x, const float* __restrict__ s,
                 const float* __restrict__ b, float* __restrict__ y)
{
    __shared__ float red[W_];
    const int row = blockIdx.x;
    const int t   = threadIdx.x;
    const float v = x[(size_t)row * W_ + t];
    red[t] = v; __syncthreads();
    for (int o = W_ / 2; o > 0; o >>= 1) {
        if (t < o) red[t] += red[t + o];
        __syncthreads();
    }
    const float mean = red[0] * (1.0f / W_);
    __syncthreads();
    const float d = v - mean;
    red[t] = d * d; __syncthreads();
    for (int o = W_ / 2; o > 0; o >>= 1) {
        if (t < o) red[t] += red[t + o];
        __syncthreads();
    }
    const float var = red[0] * (1.0f / W_);
    y[(size_t)row * W_ + t] = d * rsqrtf(var + 1e-5f) * s[t] + b[t];
}

// ---------------------------------------------------------------------------
// Fused view cross-attention:
//   pp = ray_diff @ fp_w + fp_b (K=4, inlined)
//   a  = softmax_v(kk - qq + pp);  o = sum_v a * vv
// ---------------------------------------------------------------------------
__global__ void cross_attn_kernel(const _Float16* __restrict__ kk,
                                  const _Float16* __restrict__ vv,
                                  const float* __restrict__ qq,
                                  const float* __restrict__ ray_diff,
                                  const float* __restrict__ fp_w,
                                  const float* __restrict__ fp_b,
                                  float* __restrict__ out)
{
    const int idx = blockIdx.x * blockDim.x + threadIdx.x;
    if (idx >= RS_ * W_) return;
    const int rs = idx / W_, w = idx % W_;
    const float qv = qq[idx];
    float lg[V_], mx = -3.4e38f;
#pragma unroll
    for (int v = 0; v < V_; ++v) {
        const size_t r = (size_t)rs * V_ + v;
        float pp = fp_b[w];
        const float* rd = ray_diff + r * 4;
#pragma unroll
        for (int j = 0; j < 4; ++j) pp += rd[j] * fp_w[j * W_ + w];
        const float l = (float)kk[r * W_ + w] - qv + pp;
        lg[v] = l; mx = fmaxf(mx, l);
    }
    float se = 0.0f;
#pragma unroll
    for (int v = 0; v < V_; ++v) { lg[v] = expf(lg[v] - mx); se += lg[v]; }
    float o = 0.0f;
#pragma unroll
    for (int v = 0; v < V_; ++v)
        o += lg[v] * (float)vv[((size_t)rs * V_ + v) * W_ + w];
    out[idx] = o / se;
}

// ---------------------------------------------------------------------------
// Ray self-attention (head dim 16 for q/k, 64 for v); softmax over HEADS axis
// (torch legacy nn.Softmax() dim=1 quirk reproduced).
// ---------------------------------------------------------------------------
__global__ void sattn_logits_kernel(const float* __restrict__ Q,
                                    const float* __restrict__ Kt,
                                    float* __restrict__ attn)
{
    const int idx = blockIdx.x * blockDim.x + threadIdx.x;
    if (idx >= R_ * H_ * S_ * S_) return;
    const int m = idx % S_;
    const int l = (idx / S_) % S_;
    const int h = (idx / (S_ * S_)) % H_;
    const int b = idx / (S_ * S_ * H_);
    const float* qp = Q  + ((size_t)b * S_ + l) * QK_ + h * HD_;
    const float* kp = Kt + ((size_t)b * S_ + m) * QK_ + h * HD_;
    float acc = 0.0f;
#pragma unroll
    for (int d = 0; d < HD_; ++d) acc += qp[d] * kp[d];
    attn[idx] = acc * 0.125f;   // / sqrt(W/H) = / 8
}

__global__ void sattn_softmax_h_kernel(float* __restrict__ attn)
{
    const int idx = blockIdx.x * blockDim.x + threadIdx.x;
    if (idx >= R_ * S_ * S_) return;
    const int m = idx % S_;
    const int l = (idx / S_) % S_;
    const int b = idx / (S_ * S_);
    const size_t base = ((size_t)b * H_) * (S_ * S_) + (size_t)l * S_ + m;
    const size_t str  = (size_t)S_ * S_;
    float v[H_], mx = -3.4e38f;
#pragma unroll
    for (int h = 0; h < H_; ++h) { v[h] = attn[base + h * str]; mx = fmaxf(mx, v[h]); }
    float se = 0.0f;
#pragma unroll
    for (int h = 0; h < H_; ++h) { v[h] = expf(v[h] - mx); se += v[h]; }
    const float inv = 1.0f / se;
#pragma unroll
    for (int h = 0; h < H_; ++h) attn[base + h * str] = v[h] * inv;
}

__global__ void sattn_out_kernel(const float* __restrict__ attn,
                                 const float* __restrict__ Vm,
                                 float* __restrict__ out)
{
    const int idx = blockIdx.x * blockDim.x + threadIdx.x;
    if (idx >= R_ * H_ * S_ * VD_) return;
    const int e = idx % VD_;
    const int l = (idx / VD_) % S_;
    const int h = (idx / (VD_ * S_)) % H_;
    const int b = idx / (VD_ * S_ * H_);
    const float* ap = attn + (((size_t)b * H_ + h) * S_ + l) * S_;
    float acc = 0.0f;
    for (int m = 0; m < S_; ++m)
        acc += ap[m] * Vm[((size_t)b * S_ + m) * W_ + h * VD_ + e];
    out[((size_t)b * S_ + l) * W_ + h * VD_ + e] = acc;
}

// ---------------------------------------------------------------------------
// Positional embedding: [x, sin(x*2^f), cos(x*2^f)] f=0..9  -> 63 floats
// ---------------------------------------------------------------------------
__global__ void embed_kernel(const float* __restrict__ in, float* __restrict__ out,
                             int n, int do_norm)
{
    const int i = blockIdx.x * blockDim.x + threadIdx.x;
    if (i >= n) return;
    float x0 = in[i * 3 + 0], x1 = in[i * 3 + 1], x2 = in[i * 3 + 2];
    if (do_norm) {
        const float inv = rsqrtf(x0 * x0 + x1 * x1 + x2 * x2);
        x0 *= inv; x1 *= inv; x2 *= inv;
    }
    float* o = out + (size_t)i * 63;
    o[0] = x0; o[1] = x1; o[2] = x2;
    float f = 1.0f;
#pragma unroll
    for (int k = 0; k < 10; ++k) {
        o[3 + k * 6 + 0] = sinf(x0 * f);
        o[3 + k * 6 + 1] = sinf(x1 * f);
        o[3 + k * 6 + 2] = sinf(x2 * f);
        o[3 + k * 6 + 3] = cosf(x0 * f);
        o[3 + k * 6 + 4] = cosf(x1 * f);
        o[3 + k * 6 + 5] = cosf(x2 * f);
        f *= 2.0f;
    }
}

// qc = concat([q, input_pts, input_views(broadcast per ray)], -1)  -> [RS,382]
__global__ void concat_kernel(const float* __restrict__ q,
                              const float* __restrict__ ptsE,
                              const float* __restrict__ viewE,
                              float* __restrict__ qc)
{
    const int idx = blockIdx.x * blockDim.x + threadIdx.x;
    if (idx >= RS_ * QCW_) return;
    const int rs = idx / QCW_, j = idx % QCW_;
    float v;
    if (j < W_)             v = q[(size_t)rs * W_ + j];
    else if (j < W_ + 63)   v = ptsE[(size_t)rs * 63 + (j - W_)];
    else                    v = viewE[(size_t)(rs / S_) * 63 + (j - W_ - 63)];
    qc[idx] = v;
}

// mean over samples: [R,S,W] -> [R,W]
__global__ void mean_s_kernel(const float* __restrict__ x, float* __restrict__ out)
{
    const int idx = blockIdx.x * blockDim.x + threadIdx.x;
    if (idx >= R_ * W_) return;
    const int r = idx / W_, w = idx % W_;
    float acc = 0.0f;
    for (int s = 0; s < S_; ++s) acc += x[((size_t)r * S_ + s) * W_ + w];
    out[idx] = acc * (1.0f / S_);
}

// ---------------------------------------------------------------------------
// Host orchestration
// ---------------------------------------------------------------------------
template <typename TA, typename TC>
static void gemm(hipStream_t s, const void* A, const float* B, const float* bias,
                 const float* resid, void* C, int M, int N, int K, int relu)
{
    dim3 grid((unsigned)((N + 63) / 64), (unsigned)((M + 63) / 64));
    const bool al = (M % 64 == 0) && (N % 64 == 0) && (K % 32 == 0);
    if (al)
        gemm_wmma_kernel<TA, TC, true ><<<grid, 256, 0, s>>>(
            (const TA*)A, B, bias, resid, (TC*)C, M, N, K, relu);
    else
        gemm_wmma_kernel<TA, TC, false><<<grid, 256, 0, s>>>(
            (const TA*)A, B, bias, resid, (TC*)C, M, N, K, relu);
}

static inline int cdiv(int a, int b) { return (a + b - 1) / b; }

extern "C" void kernel_launch(void* const* d_in, const int* in_sizes, int n_in,
                              void* d_out, int out_size, void* d_ws, size_t ws_size,
                              hipStream_t stream)
{
    (void)in_sizes; (void)n_in; (void)out_size; (void)ws_size;

    const float* rgb_feat  = (const float*)d_in[0];
    const float* ray_diff  = (const float*)d_in[1];
    /* mask d_in[2] is all-ones and unused by the reference math */
    const float* pts       = (const float*)d_in[3];
    const float* ray_d     = (const float*)d_in[4];
    const float* rf1_w     = (const float*)d_in[5];
    const float* rf1_b     = (const float*)d_in[6];
    const float* rf2_w     = (const float*)d_in[7];
    const float* rf2_b     = (const float*)d_in[8];
    const float* ct_norm_s = (const float*)d_in[9];
    const float* ct_norm_b = (const float*)d_in[10];
    const float* fq_w      = (const float*)d_in[11];
    const float* fq_b      = (const float*)d_in[12];
    const float* fk_w      = (const float*)d_in[13];
    const float* fk_b      = (const float*)d_in[14];
    const float* fv_w      = (const float*)d_in[15];
    const float* fv_b      = (const float*)d_in[16];
    const float* fp_w      = (const float*)d_in[17];
    const float* fp_b      = (const float*)d_in[18];
    const float* fo_w      = (const float*)d_in[19];
    const float* fo_b      = (const float*)d_in[20];
    const float* ct_fn_s   = (const float*)d_in[21];
    const float* ct_fn_b   = (const float*)d_in[22];
    const float* ffn1_w    = (const float*)d_in[23];
    const float* ffn1_b    = (const float*)d_in[24];
    const float* ffn2_w    = (const float*)d_in[25];
    const float* ffn2_b    = (const float*)d_in[26];
    const float* qfc1_w    = (const float*)d_in[27];
    const float* qfc1_b    = (const float*)d_in[28];
    const float* qfc2_w    = (const float*)d_in[29];
    const float* qfc2_b    = (const float*)d_in[30];
    const float* st_ln1_s  = (const float*)d_in[31];
    const float* st_ln1_b  = (const float*)d_in[32];
    const float* stq_w     = (const float*)d_in[33];
    const float* stq_b     = (const float*)d_in[34];
    const float* stk_w     = (const float*)d_in[35];
    const float* stk_b     = (const float*)d_in[36];
    const float* stv_w     = (const float*)d_in[37];
    const float* stv_b     = (const float*)d_in[38];
    const float* st_ln2_s  = (const float*)d_in[39];
    const float* st_ln2_b  = (const float*)d_in[40];
    const float* stf1_w    = (const float*)d_in[41];
    const float* stf1_b    = (const float*)d_in[42];
    const float* stf2_w    = (const float*)d_in[43];
    const float* stf2_b    = (const float*)d_in[44];
    const float* norm_s    = (const float*)d_in[45];
    const float* norm_b    = (const float*)d_in[46];
    const float* rgb_w     = (const float*)d_in[47];
    const float* rgb_b     = (const float*)d_in[48];

    // ---- workspace bump allocator ----
    char* wp = (char*)d_ws;
    auto alloc = [&wp](size_t bytes) -> void* {
        void* r = (void*)wp;
        wp += (bytes + 255) & ~(size_t)255;
        return r;
    };
    _Float16* featH  = (_Float16*)alloc((size_t)RSV_ * W_ * 2);   // 67 MB
    _Float16* kkH    = (_Float16*)alloc((size_t)RSV_ * W_ * 2);   // 67 MB (also MLP hidden)
    _Float16* vvH    = (_Float16*)alloc((size_t)RSV_ * W_ * 2);   // 67 MB
    _Float16* ffnH   = (_Float16*)alloc((size_t)RS_ * FF_ * 2);   // 33 MB
    float*    qbuf   = (float*)alloc((size_t)RS_ * W_ * 4);       // residual stream
    float*    xbuf   = (float*)alloc((size_t)RS_ * W_ * 4);       // layernorm out
    float*    qqbuf  = (float*)alloc((size_t)RS_ * W_ * 4);
    float*    obuf   = (float*)alloc((size_t)RS_ * W_ * 4);
    float*    qcbuf  = (float*)alloc((size_t)RS_ * QCW_ * 4);
    float*    ptsE   = (float*)alloc((size_t)RS_ * 63 * 4);
    float*    viewE  = (float*)alloc((size_t)R_ * 63 * 4);
    float*    Qh     = (float*)alloc((size_t)RS_ * QK_ * 4);
    float*    Kh     = (float*)alloc((size_t)RS_ * QK_ * 4);
    float*    Vm     = (float*)alloc((size_t)RS_ * W_ * 4);
    float*    attnB  = (float*)alloc((size_t)R_ * H_ * S_ * S_ * 4);
    float*    aoB    = (float*)alloc((size_t)RS_ * W_ * 4);
    float*    hmean  = (float*)alloc((size_t)R_ * W_ * 4);
    float*    out    = (float*)d_out;                             // [R,3] f32

    const int TPB = 256;

    // ---- positional / view embeddings ----
    embed_kernel<<<cdiv(RS_, TPB), TPB, 0, stream>>>(pts, ptsE, RS_, 0);
    embed_kernel<<<cdiv(R_, TPB), TPB, 0, stream>>>(ray_d, viewE, R_, 1);

    // ---- per-view feature MLP: feat = relu(rgb_feat@rf1+b)@rf2+b ----
    gemm<float,    _Float16>(stream, rgb_feat, rf1_w, rf1_b, nullptr, kkH,   RSV_, W_, FEAT_, 1);
    gemm<_Float16, _Float16>(stream, kkH,      rf2_w, rf2_b, nullptr, featH, RSV_, W_, W_,    0);
    rowmax_views_kernel<<<cdiv(RS_ * W_, TPB), TPB, 0, stream>>>(featH, qbuf);

    for (int i = 0; i < D_; ++i) {
        const size_t oWW = (size_t)i * W_ * W_;
        const size_t oW  = (size_t)i * W_;

        // ---- view cross-attention block ----
        layernorm_kernel<<<RS_, W_, 0, stream>>>(qbuf, ct_norm_s + oW, ct_norm_b + oW, xbuf);
        gemm<float,    float>   (stream, xbuf,  fq_w + oWW, fq_b + oW, nullptr, qqbuf, RS_,  W_, W_, 0);
        gemm<_Float16, _Float16>(stream, featH, fk_w + oWW, fk_b + oW, nullptr, kkH,   RSV_, W_, W_, 0);
        gemm<_Float16, _Float16>(stream, kkH,   fv_w + oWW, fv_b + oW, nullptr, vvH,   RSV_, W_, W_, 0);
        cross_attn_kernel<<<cdiv(RS_ * W_, TPB), TPB, 0, stream>>>(
            kkH, vvH, qqbuf, ray_diff, fp_w + (size_t)i * 4 * W_, fp_b + oW, obuf);
        gemm<float, float>(stream, obuf, fo_w + oWW, fo_b + oW, qbuf, qbuf, RS_, W_, W_, 0);

        layernorm_kernel<<<RS_, W_, 0, stream>>>(qbuf, ct_fn_s + oW, ct_fn_b + oW, xbuf);
        gemm<float,    _Float16>(stream, xbuf, ffn1_w + (size_t)i * W_ * FF_, ffn1_b + (size_t)i * FF_,
                                 nullptr, ffnH, RS_, FF_, W_, 1);
        gemm<_Float16, float>   (stream, ffnH, ffn2_w + (size_t)i * FF_ * W_, ffn2_b + oW,
                                 qbuf, qbuf, RS_, W_, FF_, 0);

        // ---- q_fc on even layers ----
        if ((i & 1) == 0) {
            const int j = i / 2;
            concat_kernel<<<cdiv(RS_ * QCW_, TPB), TPB, 0, stream>>>(qbuf, ptsE, viewE, qcbuf);
            gemm<float, float>(stream, qcbuf, qfc1_w + (size_t)j * QCW_ * W_, qfc1_b + (size_t)j * W_,
                               nullptr, obuf, RS_, W_, QCW_, 1);
            gemm<float, float>(stream, obuf, qfc2_w + (size_t)j * W_ * W_, qfc2_b + (size_t)j * W_,
                               nullptr, qbuf, RS_, W_, W_, 0);
        }

        // ---- ray self-attention block ----
        layernorm_kernel<<<RS_, W_, 0, stream>>>(qbuf, st_ln1_s + oW, st_ln1_b + oW, xbuf);
        gemm<float, float>(stream, xbuf, stq_w + (size_t)i * W_ * QK_, stq_b + (size_t)i * QK_,
                           nullptr, Qh, RS_, QK_, W_, 0);
        gemm<float, float>(stream, xbuf, stk_w + (size_t)i * W_ * QK_, stk_b + (size_t)i * QK_,
                           nullptr, Kh, RS_, QK_, W_, 0);
        gemm<float, float>(stream, xbuf, stv_w + oWW, stv_b + oW, nullptr, Vm, RS_, W_, W_, 0);
        sattn_logits_kernel<<<cdiv(R_ * H_ * S_ * S_, TPB), TPB, 0, stream>>>(Qh, Kh, attnB);
        sattn_softmax_h_kernel<<<cdiv(R_ * S_ * S_, TPB), TPB, 0, stream>>>(attnB);
        sattn_out_kernel<<<cdiv(R_ * H_ * S_ * VD_, TPB), TPB, 0, stream>>>(attnB, Vm, aoB);

        layernorm_kernel<<<RS_, W_, 0, stream>>>(aoB, st_ln2_s + oW, st_ln2_b + oW, xbuf);
        gemm<float,    _Float16>(stream, xbuf, stf1_w + (size_t)i * W_ * FF_, stf1_b + (size_t)i * FF_,
                                 nullptr, ffnH, RS_, FF_, W_, 1);
        gemm<_Float16, float>   (stream, ffnH, stf2_w + (size_t)i * FF_ * W_, stf2_b + oW,
                                 qbuf, qbuf, RS_, W_, FF_, 0);
    }

    // ---- final norm, mean over samples, rgb projection ----
    layernorm_kernel<<<RS_, W_, 0, stream>>>(qbuf, norm_s, norm_b, xbuf);
    mean_s_kernel<<<cdiv(R_ * W_, TPB), TPB, 0, stream>>>(xbuf, hmean);
    gemm<float, float>(stream, hmean, rgb_w, rgb_b, nullptr, out, R_, 3, W_, 0);
}
